// MultiHeadAttention_82145544503807
// MI455X (gfx1250) — matrix operand order repaired and checked by reading, not compiled, not run on hardware
//
#include <hip/hip_runtime.h>
#include <hip/hip_bf16.h>
#include <math.h>

typedef _Float16     v16h __attribute__((ext_vector_type(16)));
typedef _Float16     v8h  __attribute__((ext_vector_type(8)));
typedef _Float16     v4h  __attribute__((ext_vector_type(4)));
typedef float        v8f  __attribute__((ext_vector_type(8)));
typedef float        v4f  __attribute__((ext_vector_type(4)));
typedef unsigned int v4u  __attribute__((ext_vector_type(4)));

#define BM 128
#define BN 128
#define BK 32
#define BSTRIDE 36   // padded LDS row stride (halfs) for transposed B: bank-conflict-free stores
#define THREADS 256

template <typename T> struct V4T;
template <> struct V4T<float>    { typedef v4f type; };
template <> struct V4T<_Float16> { typedef v4h type; };

// ---------------------------------------------------------------------------
// GEMM: Y = X @ W + bias.  X: 16384x1024 (Tin), W: 1024x1024 (f32, converted
// to f16 at LDS staging), Y: 16384x1024 (Tout).  DUP: also write Y2 (tuple
// duplicate).  STREAM: non-temporal output stores.  N,K are compile-time so
// B-column loads use immediate offsets and row products become shifts; global
// pointers advance incrementally (no per-iteration v_mul_u64).
// 256 threads = 8 waves; 128x128 block tile, BK=32, double-buffered LDS,
// software-pipelined: load regs(kt+1) -> WMMA(kt) -> regs->LDS(kt+1).
// ---------------------------------------------------------------------------
template <typename Tin, typename Tout, bool DUP, bool STREAM>
__global__ __launch_bounds__(THREADS)
void gemm_wmma(const Tin* __restrict__ X, const float* __restrict__ W,
               const float* __restrict__ bias,
               Tout* __restrict__ Y, Tout* __restrict__ Y2)
{
    constexpr int N = 1024;
    constexpr int K = 1024;

    constexpr unsigned STAGE_BYTES =
        2u * (BM * BK + BN * BSTRIDE) * sizeof(_Float16);                 // ~34 KB
    constexpr unsigned C_BYTES = (unsigned)(BM * BN * sizeof(Tout));      // 32/64 KB
    __shared__ alignas(16) char smem[(STAGE_BYTES > C_BYTES) ? STAGE_BYTES : C_BYTES];
    _Float16* As = (_Float16*)smem;            // [2][BM*BK]      row-major [m][k]
    _Float16* Bs = As + 2 * BM * BK;           // [2][BN*BSTRIDE] transposed [n][k]
    Tout*     Cs = (Tout*)smem;                // epilogue alias

    const int tid   = threadIdx.x;
    const int lane  = tid & 31;
    const int wave  = tid >> 5;
    const int wm    = wave & 3;    // 0..3 : 32-row strip
    const int wn    = wave >> 2;   // 0..1 : 64-col strip
    const int m0    = blockIdx.y * BM;
    const int n0    = blockIdx.x * BN;

    const int khalf = lane >> 4;   // ISA 16-bit operand lane split
    const int lrow  = lane & 15;

    // staging thread mapping + incrementally-advanced global pointers
    // A: slot s -> 4 consecutive k of one row; B: 4 consecutive k of one col n
    int arow[4], acol[4], bn[4], bkq[4];
    const Tin*   ap[4];
    const float* bp[4];
#pragma unroll
    for (int s = 0; s < 4; ++s) {
        int ai  = s * THREADS + tid;
        arow[s] = ai >> 3;               // 8 vec4 per 32-wide row
        acol[s] = (ai & 7) * 4;
        bn[s]   = ai & 127;              // consecutive lanes -> consecutive n (coalesced)
        bkq[s]  = ai >> 7;               // 0..7 -> k-quad
        ap[s]   = X + (size_t)(m0 + arow[s]) * K + acol[s];
        bp[s]   = W + (size_t)(bkq[s] * 4) * N + n0 + bn[s];
    }

    v8f acc[2][4];
    const v8f zero8 = {0.f,0.f,0.f,0.f,0.f,0.f,0.f,0.f};
#pragma unroll
    for (int i = 0; i < 2; ++i)
#pragma unroll
        for (int j = 0; j < 4; ++j)
            acc[i][j] = zero8;

    constexpr int NK = K / BK;

    typename V4T<Tin>::type aval[4];     // staged A (registers)
    float bval[4][4];                    // staged B (registers)

    auto load_regs = [&]() {
#pragma unroll
        for (int s = 0; s < 4; ++s)
            aval[s] = *(const typename V4T<Tin>::type*)ap[s];
#pragma unroll
        for (int s = 0; s < 4; ++s)
#pragma unroll
            for (int i = 0; i < 4; ++i)
                bval[s][i] = bp[s][i * N];       // immediate offsets 0/4096/8192/12288 B
#pragma unroll
        for (int s = 0; s < 4; ++s) {
            ap[s] += BK;                         // advance to next K-step
            bp[s] += (size_t)BK * N;
        }
    };

    auto store_regs = [&](int buf) {
        _Float16* as = As + buf * (BM * BK);
        _Float16* bs = Bs + buf * (BN * BSTRIDE);
#pragma unroll
        for (int s = 0; s < 4; ++s) {
            v4h h;
            h[0] = (_Float16)(float)aval[s][0];
            h[1] = (_Float16)(float)aval[s][1];
            h[2] = (_Float16)(float)aval[s][2];
            h[3] = (_Float16)(float)aval[s][3];
            *(v4h*)&as[arow[s] * BK + acol[s]] = h;
        }
#pragma unroll
        for (int s = 0; s < 4; ++s) {
            v4h h;
            h[0] = (_Float16)bval[s][0];
            h[1] = (_Float16)bval[s][1];
            h[2] = (_Float16)bval[s][2];
            h[3] = (_Float16)bval[s][3];
            *(v4h*)&bs[bn[s] * BSTRIDE + bkq[s] * 4] = h;   // contiguous, conflict-free
        }
    };

    auto compute = [&](int buf) {
        const _Float16* as = As + buf * (BM * BK);
        const _Float16* bs = Bs + buf * (BN * BSTRIDE);
        v16h a[2], b[4];
#pragma unroll
        for (int i = 0; i < 2; ++i) {
            const _Float16* base = &as[(wm * 32 + i * 16 + lrow) * BK];
            ((v8h*)&a[i])[0] = *(const v8h*)(base + khalf * 16);      // K 0..7 / 16..23
            ((v8h*)&a[i])[1] = *(const v8h*)(base + khalf * 16 + 8);  // K 8..15 / 24..31
        }
#pragma unroll
        for (int j = 0; j < 4; ++j) {
            const _Float16* base = &bs[(wn * 64 + j * 16 + lrow) * BSTRIDE + khalf * 16];
            ((v4h*)&b[j])[0] = *(const v4h*)(base + 0);   // 8-B aligned b64 loads
            ((v4h*)&b[j])[1] = *(const v4h*)(base + 4);
            ((v4h*)&b[j])[2] = *(const v4h*)(base + 8);
            ((v4h*)&b[j])[3] = *(const v4h*)(base + 12);
        }
#pragma unroll
        for (int i = 0; i < 2; ++i)
#pragma unroll
            for (int j = 0; j < 4; ++j)
                acc[i][j] = __builtin_amdgcn_wmma_f32_16x16x32_f16(
                    false, a[i], false, b[j], (short)0, acc[i][j], false, false);
    };

    // -------- software-pipelined main loop ---------------------------------
    load_regs();
    store_regs(0);
    for (int kt = 0; kt < NK - 1; ++kt) {
        __syncthreads();              // buf kt&1 staged; other buffer free
        load_regs();                  // global loads issued before WMMAs
        compute(kt & 1);
        store_regs((kt + 1) & 1);     // waits loads only after compute
    }
    __syncthreads();
    compute((NK - 1) & 1);

    // -------- epilogue: bias add, stage C in LDS, coalesced b128 stores ----
    __syncthreads();                  // last fragment reads done (aliasing)
#pragma unroll
    for (int i = 0; i < 2; ++i) {
#pragma unroll
        for (int j = 0; j < 4; ++j) {
            int col  = wn * 64 + j * 16 + lrow;
            float bv = bias[n0 + col];
#pragma unroll
            for (int r = 0; r < 8; ++r) {
                int row = wm * 32 + i * 16 + r + 8 * khalf;   // C/D layout
                Cs[row * BN + col] = (Tout)(acc[i][j][r] + bv);
            }
        }
    }
    __syncthreads();

    constexpr int EPC = 16 / (int)sizeof(Tout);   // elems per 16-B chunk
    constexpr int NC  = (BM * BN) / EPC;
    constexpr int PT  = NC / THREADS;
    const unsigned int* Cw = (const unsigned int*)Cs;
#pragma unroll
    for (int s = 0; s < PT; ++s) {
        int c   = s * THREADS + tid;              // consecutive tids -> consecutive 16B
        int e   = c * EPC;
        int row = e / BN;
        int col = e % BN;
        v4u val = *(const v4u*)(Cw + c * 4);
        size_t o = (size_t)(m0 + row) * N + n0 + col;
        if (STREAM) __builtin_nontemporal_store(val, (v4u*)(Y + o));
        else        *(v4u*)(Y + o) = val;
        if (DUP) {
            if (STREAM) __builtin_nontemporal_store(val, (v4u*)(Y2 + o));
            else        *(v4u*)(Y2 + o) = val;
        }
    }
}

// ---------------------------------------------------------------------------
// Per-token attention over the HEAD axis (reference never transposes, so
// scores are [H,H] per token).  4 tokens per 256-thread block; 64 threads per
// token.  q/k/v are f16 in workspace; compute in f32; concat written as f16.
// ---------------------------------------------------------------------------
__global__ __launch_bounds__(256)
void attn_kernel(const _Float16* __restrict__ q, const _Float16* __restrict__ k,
                 const _Float16* __restrict__ v, const float* __restrict__ mask,
                 _Float16* __restrict__ concat)
{
    __shared__ _Float16 sq[4][1024];
    __shared__ _Float16 sk[4][1024];
    __shared__ _Float16 sv[4][1024];
    __shared__ float    sc[4][16][16];

    const int tid = threadIdx.x;
    const int tl  = tid >> 6;          // token local 0..3
    const int u   = tid & 63;
    const size_t t    = (size_t)blockIdx.x * 4 + tl;
    const size_t base = t * 1024;

#pragma unroll
    for (int s = 0; s < 4; ++s) {
        int off = u * 16 + s * 4;
        *(v4h*)&sq[tl][off] = *(const v4h*)&q[base + off];
        *(v4h*)&sk[tl][off] = *(const v4h*)&k[base + off];
        *(v4h*)&sv[tl][off] = *(const v4h*)&v[base + off];
    }
    __syncthreads();

    const int h  = u >> 2;             // head row 0..15
    const int g0 = (u & 3) * 4;        // 4 score columns per thread
    const float scale = 0.125f;        // 1/sqrt(64)

#pragma unroll
    for (int gg = 0; gg < 4; ++gg) {
        int g = g0 + gg;
        float s = 0.f;
#pragma unroll 8
        for (int d = 0; d < 64; ++d)
            s += (float)sq[tl][h * 64 + d] * (float)sk[tl][g * 64 + d];
        s = s * scale + mask[(t * 16 + h) * 16 + g] * (-1e9f);
        sc[tl][h][g] = s;
    }
    __syncthreads();

    float mx = -INFINITY;
#pragma unroll
    for (int g = 0; g < 16; ++g) mx = fmaxf(mx, sc[tl][h][g]);
    float sum = 0.f;
#pragma unroll
    for (int g = 0; g < 16; ++g) sum += __expf(sc[tl][h][g] - mx);
    float inv = 1.0f / sum;
    float w[4];
#pragma unroll
    for (int gg = 0; gg < 4; ++gg) w[gg] = __expf(sc[tl][h][g0 + gg] - mx) * inv;
    __syncthreads();                   // all reads of raw scores complete
#pragma unroll
    for (int gg = 0; gg < 4; ++gg) sc[tl][h][g0 + gg] = w[gg];
    __syncthreads();

    const int d0 = (u & 3) * 16;
#pragma unroll
    for (int d = 0; d < 16; ++d) {
        float o = 0.f;
#pragma unroll
        for (int g = 0; g < 16; ++g)
            o += sc[tl][h][g] * (float)sv[tl][g * 64 + d0 + d];
        concat[base + h * 64 + d0 + d] = (_Float16)o;
    }
}

// ---------------------------------------------------------------------------
extern "C" void kernel_launch(void* const* d_in, const int* in_sizes, int n_in,
                              void* d_out, int out_size, void* d_ws, size_t ws_size,
                              hipStream_t stream)
{
    const float* Q    = (const float*)d_in[0];
    const float* K_   = (const float*)d_in[1];
    const float* V    = (const float*)d_in[2];
    const float* mask = (const float*)d_in[3];
    const float* Wq   = (const float*)d_in[4];
    const float* bq   = (const float*)d_in[5];
    const float* Wk   = (const float*)d_in[6];
    const float* bk   = (const float*)d_in[7];
    const float* Wv   = (const float*)d_in[8];
    const float* bv   = (const float*)d_in[9];
    const float* Wo   = (const float*)d_in[10];
    const float* bo   = (const float*)d_in[11];

    const int M = 4 * 4096, N = 1024;

    _Float16* qh = (_Float16*)d_ws;              // 32 MB
    _Float16* kh = qh + (size_t)M * N;           // 32 MB
    _Float16* vh = kh + (size_t)M * N;           // 32 MB
    _Float16* ch = vh + (size_t)M * N;           // 32 MB
    float* out   = (float*)d_out;
    float* out2  = out + (size_t)M * N;          // tuple duplicate

    dim3 grid(N / BN, M / BM);
    dim3 block(THREADS);

    gemm_wmma<float, _Float16, false, false><<<grid, block, 0, stream>>>(
        Q,  Wq, bq, qh, (_Float16*)nullptr);
    gemm_wmma<float, _Float16, false, false><<<grid, block, 0, stream>>>(
        K_, Wk, bk, kh, (_Float16*)nullptr);
    gemm_wmma<float, _Float16, false, false><<<grid, block, 0, stream>>>(
        V,  Wv, bv, vh, (_Float16*)nullptr);

    attn_kernel<<<M / 4, 256, 0, stream>>>(qh, kh, vh, mask, ch);

    gemm_wmma<_Float16, float, true, true><<<grid, block, 0, stream>>>(
        ch, Wo, bo, out, out2);
}
